// snn_layer_10591389352401
// MI455X (gfx1250) — matrix operand order repaired and checked by reading, not compiled, not run on hardware
//
#include <hip/hip_runtime.h>

typedef __bf16 bf16;
typedef __attribute__((ext_vector_type(16))) __bf16 v16bf;
typedef __attribute__((ext_vector_type(8)))  float  v8f;

#define TILE_M 128
#define TILE_N 128
#define TILE_K 32
#define RS 34   // LDS row stride in bf16 elements (68 B): dword-aligned rows,
                // store lane-stride 68 dw -> gcd(68,64)=4 -> 2-way conflicts max;
                // frag-load lane-stride 17 dw -> gcd(17,64)=1 -> conflict-free.

__device__ __forceinline__ v8f v8f_zero() {
  v8f z;
#pragma unroll
  for (int i = 0; i < 8; ++i) z[i] = 0.0f;
  return z;
}

// Split f into hi (truncated bf16) + lo (RNE bf16 of remainder); return packed
// hi pair (f0 low half, f1 high half) and packed lo pair via out-param.
__device__ __forceinline__ unsigned split_pack(float f0, float f1, unsigned* lopack) {
  const unsigned u0 = __float_as_uint(f0) & 0xFFFF0000u;
  const unsigned u1 = __float_as_uint(f1) & 0xFFFF0000u;
  const float d0 = f0 - __uint_as_float(u0);
  const float d1 = f1 - __uint_as_float(u1);
  union { bf16 h[2]; unsigned u; } lp;
  lp.h[0] = (bf16)d0;
  lp.h[1] = (bf16)d1;
  *lopack = lp.u;
  return (u0 >> 16) | u1;
}

// A 16x32 bf16 fragment (ISA 7.12.2): lane<16: row=lane, K{0..7,16..23};
// lane>=16: row=lane-16, K{8..15,24..31}. LDS layout: [row][k], row stride RS.
__device__ __forceinline__ v16bf load_fragA(const bf16* base, int lane) {
  const int half = lane >> 4;
  const int m    = lane & 15;
  const unsigned* p = (const unsigned*)(base + m * RS) + half * 4;  // K base 0 or 8
  union { unsigned u[8]; v16bf v; } r;
#pragma unroll
  for (int i = 0; i < 4; ++i) r.u[i] = p[i];         // K base+0..7
#pragma unroll
  for (int i = 0; i < 4; ++i) r.u[4 + i] = p[8 + i]; // K base+16..23
  return r.v;
}

// B 32x16 bf16 fragment: lane<16: col=lane, K{0..7,16..23}; lane>=16: col=lane-16, K{8..15,24..31}.
// LDS layout: [col][k], row stride RS.
__device__ __forceinline__ v16bf load_fragB(const bf16* base, int lane) {
  const int half = lane >> 4;
  const int n    = lane & 15;
  const unsigned* p = (const unsigned*)(base + n * RS) + half * 4;
  union { unsigned u[8]; v16bf v; } r;
#pragma unroll
  for (int i = 0; i < 4; ++i) r.u[i] = p[i];
#pragma unroll
  for (int i = 0; i < 4; ++i) r.u[4 + i] = p[8 + i];
  return r.v;
}

// h[m, u] = sum_k A[m,k] * W[k,u],  A[m,k] = x[b, k, t], m = b*256 + t.
// Split-f32 GEMM: A = Ahi + Alo (bf16 each), 3 bf16 WMMAs per K-step recover ~f32 accuracy.
// Software-pipelined: tile k0+32 is loaded+converted while tile k0 feeds the WMMAs.
__global__ __launch_bounds__(256) void snn_gemm_kernel(const float* __restrict__ x,
                                                       const float* __restrict__ W,
                                                       float* __restrict__ h) {
  __shared__ __align__(16) bf16 As_hi[TILE_M * RS];
  __shared__ __align__(16) bf16 As_lo[TILE_M * RS];
  __shared__ __align__(16) bf16 Bs_hi[TILE_N * RS];
  __shared__ __align__(16) bf16 Bs_lo[TILE_N * RS];

  const int tid  = threadIdx.x;
  const int lane = tid & 31;
  const int w    = tid >> 5;     // wave 0..7
  const int wr   = w & 3;        // 32-row band within tile
  const int wc   = w >> 2;       // 64-col band within tile

  const int row0 = blockIdx.y * TILE_M;   // m tile base (aligned so one batch per tile)
  const int col0 = blockIdx.x * TILE_N;   // u tile base
  const int b    = row0 >> 8;             // batch index
  const int t0   = row0 & 255;            // time base (0 or 128)

  const float* xA = x + (size_t)b * (1024 * 256) + t0;  // A[r,k] = xA[k*256 + r] (m-contiguous)
  const float* WB = W + col0;                           // B[k,uu] = WB[k*1024 + uu]

  const int r4 = tid & 31;   // 4-row group within tile (rows r4*4 .. r4*4+3)
  const int kb = tid >> 5;   // 0..7

  // Prefetched + converted tile fragments held across the compute phase.
  unsigned hA[2][4], lA[2][4], hB[2][4], lB[2][4];

  auto prefetch_convert = [&](int k0) {
#pragma unroll
    for (int l = 0; l < 2; ++l) {
      const int kk = (kb + l * 8) * 2;  // even: 0,2,...,30
      const float4 a0 = *(const float4*)(xA + (size_t)(k0 + kk) * 256 + r4 * 4);
      const float4 a1 = *(const float4*)(xA + (size_t)(k0 + kk + 1) * 256 + r4 * 4);
      const float4 b0 = *(const float4*)(WB + (size_t)(k0 + kk) * 1024 + r4 * 4);
      const float4 b1 = *(const float4*)(WB + (size_t)(k0 + kk + 1) * 1024 + r4 * 4);
      const float a0v[4] = {a0.x, a0.y, a0.z, a0.w};
      const float a1v[4] = {a1.x, a1.y, a1.z, a1.w};
      const float b0v[4] = {b0.x, b0.y, b0.z, b0.w};
      const float b1v[4] = {b1.x, b1.y, b1.z, b1.w};
#pragma unroll
      for (int j = 0; j < 4; ++j) {
        hA[l][j] = split_pack(a0v[j], a1v[j], &lA[l][j]);
        hB[l][j] = split_pack(b0v[j], b1v[j], &lB[l][j]);
      }
    }
  };

  v8f acc[2][4];
#pragma unroll
  for (int rb = 0; rb < 2; ++rb)
#pragma unroll
    for (int cb = 0; cb < 4; ++cb) acc[rb][cb] = v8f_zero();

  prefetch_convert(0);

  for (int k0 = 0; k0 < 1024; k0 += TILE_K) {
    __syncthreads();
    // Short critical section: 32 packed b32 LDS stores (transpose into [row][k]).
#pragma unroll
    for (int l = 0; l < 2; ++l) {
      const int kk = (kb + l * 8) * 2;
#pragma unroll
      for (int j = 0; j < 4; ++j) {
        const int row = r4 * 4 + j;
        *(unsigned*)(As_hi + row * RS + kk) = hA[l][j];
        *(unsigned*)(As_lo + row * RS + kk) = lA[l][j];
        *(unsigned*)(Bs_hi + row * RS + kk) = hB[l][j];
        *(unsigned*)(Bs_lo + row * RS + kk) = lB[l][j];
      }
    }
    __syncthreads();

    // Kick off next tile's global loads + conversion; overlaps with WMMAs below.
    if (k0 + TILE_K < 1024) prefetch_convert(k0 + TILE_K);

    v16bf ah[2], al[2];
#pragma unroll
    for (int rb = 0; rb < 2; ++rb) {
      const int rbase = (wr * 32 + rb * 16) * RS;
      ah[rb] = load_fragA(&As_hi[rbase], lane);
      al[rb] = load_fragA(&As_lo[rbase], lane);
    }
    v16bf bh[4], bl[4];
#pragma unroll
    for (int cb = 0; cb < 4; ++cb) {
      const int cbase = (wc * 64 + cb * 16) * RS;
      bh[cb] = load_fragB(&Bs_hi[cbase], lane);
      bl[cb] = load_fragB(&Bs_lo[cbase], lane);
    }

#pragma unroll
    for (int rb = 0; rb < 2; ++rb)
#pragma unroll
      for (int cb = 0; cb < 4; ++cb) {
        acc[rb][cb] = __builtin_amdgcn_wmma_f32_16x16x32_bf16(
            false, ah[rb], false, bh[cb], (short)0, acc[rb][cb], false, false);
        acc[rb][cb] = __builtin_amdgcn_wmma_f32_16x16x32_bf16(
            false, ah[rb], false, bl[cb], (short)0, acc[rb][cb], false, false);
        acc[rb][cb] = __builtin_amdgcn_wmma_f32_16x16x32_bf16(
            false, al[rb], false, bh[cb], (short)0, acc[rb][cb], false, false);
      }
  }

  // C/D layout (ISA 7.12.2): lane gives N = lane&15; VGPR g holds M = g + 8*(lane>>4).
  const int n     = lane & 15;
  const int halfm = lane >> 4;
#pragma unroll
  for (int rb = 0; rb < 2; ++rb) {
    const int mbase = row0 + wr * 32 + rb * 16 + halfm * 8;
#pragma unroll
    for (int cb = 0; cb < 4; ++cb) {
      const int ug = col0 + wc * 64 + cb * 16 + n;
#pragma unroll
      for (int g = 0; g < 8; ++g) {
        h[(size_t)(mbase + g) * 1024 + ug] = acc[rb][cb][g];
      }
    }
  }
}

// Sequential LIF scan over t per (b,u). h is [b,t,u]; out is [t,b,u].
__global__ __launch_bounds__(256) void snn_scan_kernel(const float* __restrict__ h,
                                                       float* __restrict__ out) {
  const int g = blockIdx.x * 256 + threadIdx.x;  // g = b*1024 + u
  const int b = g >> 10;
  const int u = g & 1023;
  const float* hrow = h + (size_t)b * (256 * 1024) + u;
  float* orow = out + g;                          // out[t*131072 + g]
  float v = 0.0f, syn = 0.0f;
#pragma unroll 4
  for (int t = 0; t < 256; ++t) {
    const float hstep = hrow[(size_t)t * 1024];
    const float s = 1.0f / (1.0f + __expf(1.0f - v));   // sigmoid(v - THRESH)
    __builtin_nontemporal_store(s, &orow[(size_t)t * 131072]);  // out never re-read
    const float vn = fmaf(0.85f, v, syn) * (1.0f - s);  // (beta*v + syn)*(1 - s)
    syn = fmaf(0.9f, syn, hstep);                       // alpha*syn + h
    v = vn;
  }
}

extern "C" void kernel_launch(void* const* d_in, const int* in_sizes, int n_in,
                              void* d_out, int out_size, void* d_ws, size_t ws_size,
                              hipStream_t stream) {
  const float* x = (const float*)d_in[0];   // [128, 1024, 256] f32
  const float* W = (const float*)d_in[1];   // [1024, 1024] f32
  float* out = (float*)d_out;               // [256, 128, 1024] f32
  float* h   = (float*)d_ws;                // [32768, 1024] f32 = 128 MiB scratch

  dim3 ggrid(1024 / TILE_N, 32768 / TILE_M, 1);  // (8, 256)
  snn_gemm_kernel<<<ggrid, 256, 0, stream>>>(x, W, h);

  snn_scan_kernel<<<(128 * 1024) / 256, 256, 0, stream>>>(h, out);
}
